// MargeSimpson_47588237639853
// MI455X (gfx1250) — matrix operand order repaired and checked by reading, not compiled, not run on hardware
//
#include <hip/hip_runtime.h>

typedef __attribute__((ext_vector_type(8)))  _Float16 v8h;
typedef __attribute__((ext_vector_type(16))) _Float16 v16h;
typedef __attribute__((ext_vector_type(8)))  float    v8f;

#define B_     64
#define H_     128
#define W_     2048
#define UNITS_ 32
#define VOCAB_ 201
#define NGATE  128   // 4*UNITS
#define NCOL   256   // both directions concatenated

// ---------------------------------------------------------------------------
// Pack Wf|Wb ([128,128] each) and Uf/Ub ([32,128]) into f16 B-operand
// fragments (32 lanes x 16 halves, contiguous per lane), plus bias concat.
// B fragment mapping (16x16x32 f16 WMMA): lane L -> N = n0 + (L&15),
// slot i -> K = 16*(L>>4) + i.
// ---------------------------------------------------------------------------
__global__ void pack_weights_kernel(const float* __restrict__ Wf,
                                    const float* __restrict__ Wb,
                                    const float* __restrict__ Uf,
                                    const float* __restrict__ Ub,
                                    const float* __restrict__ bf,
                                    const float* __restrict__ bb,
                                    _Float16* __restrict__ WfbPack,
                                    _Float16* __restrict__ UPack,
                                    float* __restrict__ biasFB) {
  int idx = blockIdx.x * blockDim.x + threadIdx.x;
  if (idx < 4 * 16 * 512) {              // 4 k-tiles x 16 n-tiles
    int frag = idx >> 9, within = idx & 511;
    int kt = frag >> 4, nt = frag & 15;
    int L = within >> 4, i = within & 15;
    int k = kt * 32 + ((L >> 4) << 4) + i;
    int n = nt * 16 + (L & 15);
    float v = (n < NGATE) ? Wf[k * NGATE + n] : Wb[k * NGATE + (n - NGATE)];
    WfbPack[idx] = (_Float16)v;
    return;
  }
  int idx2 = idx - 4 * 16 * 512;
  if (idx2 < 2 * 8 * 512) {              // 2 dirs x 8 n-tiles, K=32
    int dir = idx2 >> 12;
    int rem = idx2 & 4095;
    int nt = rem >> 9, within = rem & 511;
    int L = within >> 4, i = within & 15;
    int k = ((L >> 4) << 4) + i;
    int n = nt * 16 + (L & 15);
    const float* U = dir ? Ub : Uf;
    UPack[idx2] = (_Float16)U[k * NGATE + n];
    return;
  }
  int idx3 = idx2 - 2 * 8 * 512;
  if (idx3 < NCOL) biasFB[idx3] = (idx3 < NGATE) ? bf[idx3] : bb[idx3 - NGATE];
}

// ---------------------------------------------------------------------------
// 5x5 SAME conv (no bias) fused with the [B,H,W] -> [B,W,H] transpose; emits
// the GEMM A matrix in f16. One block = (b, 32 w-columns), LDS tile transpose.
// ---------------------------------------------------------------------------
__global__ void conv_transpose_kernel(const float* __restrict__ x,
                                      const float* __restrict__ ck,
                                      _Float16* __restrict__ seqA) {
  __shared__ __attribute__((aligned(32))) _Float16 tile[32 * 130];
  int b = blockIdx.x;
  int w0 = blockIdx.y * 32;
  int wt = threadIdx.x & 31;
  int hg = threadIdx.x >> 5;
  float kk[25];
#pragma unroll
  for (int i = 0; i < 25; i++) kk[i] = ck[i];
  int w = w0 + wt;
  for (int hh = hg; hh < H_; hh += 8) {
    float acc = 0.f;
#pragma unroll
    for (int dh = 0; dh < 5; dh++) {
      int h2 = hh + dh - 2;
      if ((unsigned)h2 < (unsigned)H_) {
        const float* row = x + ((size_t)(b * H_ + h2)) * W_;
#pragma unroll
        for (int dw = 0; dw < 5; dw++) {
          int w2 = w + dw - 2;
          if ((unsigned)w2 < (unsigned)W_) acc += row[w2] * kk[dh * 5 + dw];
        }
      }
    }
    tile[wt * 130 + hh] = (_Float16)acc;
  }
  __syncthreads();
  for (int i = threadIdx.x; i < 32 * H_; i += 256) {
    int ww = i >> 7, h = i & 127;
    seqA[((size_t)(b * W_ + w0 + ww)) * H_ + h] = tile[ww * 130 + h];
  }
}

// ---------------------------------------------------------------------------
// xw = seqA[131072,128] @ (Wf|Wb)[128,256] + bias, f16 WMMA, f32 accumulate.
// Block = 64 rows x 256 cols; 8 waves, each wave: 1 M-tile x 8 N-tiles.
// ---------------------------------------------------------------------------
__global__ void xw_gemm_kernel(const _Float16* __restrict__ seqA,
                               const _Float16* __restrict__ WfbPack,
                               const float* __restrict__ biasFB,
                               float* __restrict__ xw) {
  int tid = threadIdx.x;
  int wave = tid >> 5, L = tid & 31;
  int half = L >> 4, lm = L & 15;
  int m_t = wave & 3;
  int nb = (wave >> 2) * 8;                       // first n-tile of this wave
  size_t m0 = (size_t)blockIdx.x * 64 + (size_t)m_t * 16;
  v8f acc[8] = {};
  const _Float16* arow = seqA + (m0 + lm) * 128;  // lane's A row (M = lm)
#pragma unroll
  for (int kt = 0; kt < 4; kt++) {
    const _Float16* abase = arow + kt * 32 + half * 16;
    __builtin_prefetch(abase + 64 * 128, 0, 1);   // next block's rows -> global_prefetch_b8
    v16h a = *(const v16h*)abase;                 // K = kt*32 + 16*half + 0..15
#pragma unroll
    for (int j = 0; j < 8; j++) {
      int nt = nb + j;
      v16h bfrag = *(const v16h*)(WfbPack + (((kt * 16 + nt) * 32) + L) * 16);
      acc[j] = __builtin_amdgcn_wmma_f32_16x16x32_f16(
          false, a, false, bfrag, (short)0, acc[j], false, false);
    }
  }
#pragma unroll
  for (int j = 0; j < 8; j++) {
    int n0 = (nb + j) * 16;
#pragma unroll
    for (int r = 0; r < 8; r++) {
      size_t row = m0 + r + 8 * half;             // D layout: M = r + 8*(L>=16)
      int col = n0 + lm;
      xw[row * NCOL + col] = acc[j][r] + biasFB[col];
    }
  }
}

// ---------------------------------------------------------------------------
// LSTM recurrence. grid = 8 blocks: dir (2) x batch-group of 16 (4).
// Per step: gates[16,128] = h16[16,32] @ U[32,128] + xw  via ONE WMMA per
// wave (wave = N-tile, U fragment kept in VGPRs for all 2048 steps).
// The xw C-fragment for step t+1 is software-pipelined: its 8 global loads
// issue before step t's WMMA and complete under the elementwise/barrier work,
// keeping only  barrier -> LDS h -> WMMA -> LDS gates  on the serial path.
// Backward dir reads xw at the ragged-reversed time index.
// ---------------------------------------------------------------------------
__global__ void lstm_kernel(const float* __restrict__ xw,
                            const _Float16* __restrict__ UPack,
                            const int* __restrict__ widths,
                            float* __restrict__ outF,
                            float* __restrict__ outBrev) {
  __shared__ __attribute__((aligned(32))) _Float16 h16[16 * 32];
  __shared__ float gates[16 * NGATE];
  __shared__ int wds[16];
  int dir = blockIdx.x >> 2;
  int bg = blockIdx.x & 3;
  int tid = threadIdx.x;
  int wave = tid >> 5, L = tid & 31;
  int half = L >> 4, lm = L & 15;
  float* outp = dir ? outBrev : outF;

  if (tid < 16) wds[tid] = widths[bg * 16 + tid];
  for (int i = tid; i < 16 * 32; i += 256) h16[i] = (_Float16)0.f;
  // persistent B operand: U fragment for this wave's N-tile
  v16h uf = *(const v16h*)(UPack + (((dir * 8 + wave) * 32) + L) * 16);
  // cell state lives in registers: this thread owns entries tid*2, tid*2+1
  float cst[2] = {0.f, 0.f};
  int em[2], eu[2];
#pragma unroll
  for (int j = 0; j < 2; j++) { int e = tid * 2 + j; em[j] = e >> 5; eu[j] = e & 31; }
  __syncthreads();

  int n = wave * 16 + lm;
  // per-lane row bases and width for the 8 C-fragment rows
  size_t rbase[8];
  int rwd[8];
#pragma unroll
  for (int r = 0; r < 8; r++) {
    int m = r + 8 * half;
    int b = bg * 16 + m;
    rwd[r] = wds[m];
    rbase[r] = ((size_t)b * W_) * NCOL + dir * NGATE + n;
  }
  // preload C fragment for t = 0
  v8f cnext;
#pragma unroll
  for (int r = 0; r < 8; r++) {
    int te0 = (dir && 0 < rwd[r]) ? (rwd[r] - 1) : 0;
    cnext[r] = xw[rbase[r] + (size_t)te0 * NCOL];
  }

  for (int t = 0; t < W_; t++) {
    v8f c = cnext;
    // A fragment from h16 (lane M = lm, K-halves by lane group)
    v16h a = *(const v16h*)&h16[lm * 32 + half * 16];
    // issue next step's xw loads now; they retire under the WMMA + cell math
    if (t + 1 < W_) {
      int tn = t + 1;
#pragma unroll
      for (int r = 0; r < 8; r++) {
        int te = (dir && tn < rwd[r]) ? (rwd[r] - 1 - tn) : tn;
        cnext[r] = xw[rbase[r] + (size_t)te * NCOL];
      }
    }
    v8f d = __builtin_amdgcn_wmma_f32_16x16x32_f16(
        false, a, false, uf, (short)0, c, false, false);
#pragma unroll
    for (int r = 0; r < 8; r++) gates[(r + 8 * half) * NGATE + n] = d[r];
    __syncthreads();
    // elementwise LSTM cell update (keras gate order i,f,g,o)
#pragma unroll
    for (int j = 0; j < 2; j++) {
      int m = em[j], u = eu[j];
      float zi = gates[m * NGATE + u];
      float zf = gates[m * NGATE + 32 + u];
      float zg = gates[m * NGATE + 64 + u];
      float zo = gates[m * NGATE + 96 + u];
      float ig = 1.f / (1.f + __expf(-zi));
      float fg = 1.f / (1.f + __expf(-zf));
      float gg = tanhf(zg);
      float og = 1.f / (1.f + __expf(-zo));
      float cc = fg * cst[j] + ig * gg;
      cst[j] = cc;
      float hh = og * tanhf(cc);
      h16[m * 32 + u] = (_Float16)hh;
      int b = bg * 16 + m;
      outp[((size_t)b * W_ + t) * UNITS_ + u] = hh;
    }
    __syncthreads();
  }
}

// ---------------------------------------------------------------------------
// merged = out_f + gather(out_b_rev); Dense[32->201] + softmax + ragged mask.
// One wave per (b,t) row; 105 MB output write dominates -> VALU is fine here.
// ---------------------------------------------------------------------------
__global__ void dense_softmax_kernel(const float* __restrict__ outF,
                                     const float* __restrict__ outBrev,
                                     const float* __restrict__ Wd,
                                     const float* __restrict__ bd,
                                     const int* __restrict__ widths,
                                     float* __restrict__ out) {
  int wave = threadIdx.x >> 5, lane = threadIdx.x & 31;
  size_t row = (size_t)blockIdx.x * 8 + wave;
  int b = (int)(row >> 11);   // / 2048
  int t = (int)(row & 2047);
  int wd = widths[b];
  float* dst = out + row * VOCAB_;
  if (t >= wd) {
#pragma unroll
    for (int j = 0; j < 7; j++) { int nn = lane + 32 * j; if (nn < VOCAB_) dst[nn] = 0.f; }
    return;
  }
  int rv = wd - 1 - t;
  float mv = outF[row * UNITS_ + lane] +
             outBrev[((size_t)b * W_ + rv) * UNITS_ + lane];
  float acc[7];
#pragma unroll
  for (int j = 0; j < 7; j++) { int nn = lane + 32 * j; acc[j] = (nn < VOCAB_) ? bd[nn] : -1e30f; }
#pragma unroll
  for (int u = 0; u < 32; u++) {
    float mu = __shfl(mv, u, 32);
#pragma unroll
    for (int j = 0; j < 7; j++) {
      int nn = lane + 32 * j;
      if (nn < VOCAB_) acc[j] += mu * Wd[u * VOCAB_ + nn];
    }
  }
  float mx = -1e30f;
#pragma unroll
  for (int j = 0; j < 7; j++) { int nn = lane + 32 * j; if (nn < VOCAB_) mx = fmaxf(mx, acc[j]); }
#pragma unroll
  for (int o = 16; o > 0; o >>= 1) mx = fmaxf(mx, __shfl_xor(mx, o, 32));
  float s = 0.f;
#pragma unroll
  for (int j = 0; j < 7; j++) {
    int nn = lane + 32 * j;
    if (nn < VOCAB_) { acc[j] = __expf(acc[j] - mx); s += acc[j]; }
  }
#pragma unroll
  for (int o = 16; o > 0; o >>= 1) s += __shfl_xor(s, o, 32);
  float inv = 1.f / s;
#pragma unroll
  for (int j = 0; j < 7; j++) { int nn = lane + 32 * j; if (nn < VOCAB_) dst[nn] = acc[j] * inv; }
}

// ---------------------------------------------------------------------------
// Workspace layout (bytes):
//   seqA    f16 [131072,128]          @ 0           33,554,432
//   xw      f32 [131072,256]          @ 33,554,432  134,217,728  (fits in 192MB L2)
//   outF    f32 [64,2048,32]          @ 167,772,160 16,777,216
//   outBrev f32 [64,2048,32]          @ 184,549,376 16,777,216
//   WfbPack f16 fragments             @ 201,326,592 65,536
//   UPack   f16 fragments             @ 201,392,128 16,384
//   biasFB  f32 [256]                 @ 201,408,512 1,024
// ---------------------------------------------------------------------------
extern "C" void kernel_launch(void* const* d_in, const int* in_sizes, int n_in,
                              void* d_out, int out_size, void* d_ws, size_t ws_size,
                              hipStream_t stream) {
  (void)in_sizes; (void)n_in; (void)out_size; (void)ws_size;
  const float* x  = (const float*)d_in[0];
  const float* ck = (const float*)d_in[1];
  const float* Wf = (const float*)d_in[2];
  const float* Uf = (const float*)d_in[3];
  const float* bf = (const float*)d_in[4];
  const float* Wb = (const float*)d_in[5];
  const float* Ub = (const float*)d_in[6];
  const float* bb = (const float*)d_in[7];
  const float* Wd = (const float*)d_in[8];
  const float* bd = (const float*)d_in[9];
  const int* widths = (const int*)d_in[10];

  char* ws = (char*)d_ws;
  _Float16* seqA   = (_Float16*)(ws);
  float*    xw     = (float*)(ws + 33554432);
  float*    outF   = (float*)(ws + 167772160);
  float*    outBr  = (float*)(ws + 184549376);
  _Float16* WfbP   = (_Float16*)(ws + 201326592);
  _Float16* UP     = (_Float16*)(ws + 201392128);
  float*    biasFB = (float*)(ws + 201408512);
  float* out = (float*)d_out;

  pack_weights_kernel<<<161, 256, 0, stream>>>(Wf, Wb, Uf, Ub, bf, bb, WfbP, UP, biasFB);
  conv_transpose_kernel<<<dim3(B_, W_ / 32), 256, 0, stream>>>(x, ck, seqA);
  xw_gemm_kernel<<<(B_ * W_) / 64, 256, 0, stream>>>(seqA, WfbP, biasFB, xw);
  lstm_kernel<<<8, 256, 0, stream>>>(xw, UP, widths, outF, outBr);
  dense_softmax_kernel<<<(B_ * W_) / 8, 256, 0, stream>>>(outF, outBr, Wd, bd, widths, out);
}
